// RouteMoELayer_26190710571114
// MI455X (gfx1250) — matrix operand order repaired and checked by reading, not compiled, not run on hardware
//
#include <hip/hip_runtime.h>
#include <hip/hip_bf16.h>
#include <math.h>

// Problem constants (from reference)
#define BZ   128
#define T    32
#define D    768
#define DFF  3072
#define E    8
#define NROWS 256   // BZ * BEAMS

typedef __attribute__((ext_vector_type(16))) __bf16 v16bf;
typedef __attribute__((ext_vector_type(8)))  float  v8f;

union Frag {
    v16bf v;
    uint4 q[2];
    unsigned short u[16];
};

__device__ __forceinline__ unsigned short f2bf(float f) {
    union { __bf16 h; unsigned short u; } c;
    c.h = (__bf16)f;            // hardware cvt, RNE
    return c.u;
}

__device__ __forceinline__ unsigned int pk2bf(float lo, float hi) {
    union { __bf16 h[2]; unsigned int u; } c;
    c.h[0] = (__bf16)lo; c.h[1] = (__bf16)hi;
    return c.u;
}

// Branch-free exact-form gelu: 0.5*x*(1+erf(x/sqrt(2))) with erf via the
// Abramowitz-Stegun 7.1.26 rational approximation (|err| < 1.5e-7, i.e.
// fp32-accurate) -- 4 FMAs + v_rcp_f32 + v_exp_f32, no EXEC divergence
// (TRANS ops co-execute with WMMA).
__device__ __forceinline__ float gelu_erf(float v) {
    float xs = v * 0.70710678118654752f;
    float ax = fabsf(xs);
    float t  = __builtin_amdgcn_rcpf(fmaf(0.3275911f, ax, 1.0f));
    float p  = fmaf(t, 1.061405429f, -1.453152027f);
    p = fmaf(t, p, 1.421413741f);
    p = fmaf(t, p, -0.284496736f);
    p = fmaf(t, p, 0.254829592f);
    float ex = __expf(-ax * ax);
    float er = copysignf(fmaf(-p * t, ex, 1.0f), xs);
    return 0.5f * v * (1.0f + er);
}

// ---------------------------------------------------------------------------
// K0: one-time fp32 -> bf16 conversion of W1/W2 PLUS swizzle into WMMA
// B-fragment tile order.  Logical weight (K_total x N_total) row-major is
// stored as 32(K) x 16(N) tiles; within a tile, lane l = kh*16 + nc owns the
// 16 elements {K = kt*32 + (j&8)*2 + kh*8 + (j&7), N = nt*16 + nc} stored contiguously
// (32 bytes).  Tile order: n-tile major, k-tile minor (sequential k-loop).
// A B fragment in the GEMM is then exactly two global_load_b128.
// ---------------------------------------------------------------------------
__global__ void k_swizzle(const float* __restrict__ w1,
                          const float* __restrict__ w2,
                          unsigned short* __restrict__ o1,
                          unsigned short* __restrict__ o2) {
    __shared__ unsigned short lt[32 * 128];   // 8 KB chunk buffer

    const int cid = blockIdx.x;               // 0 .. 2*E*576-1
    const float* src; unsigned short* dst; int Kt, Nt; int local;
    if (cid < E * 576) { src = w1; dst = o1; Kt = D;   Nt = DFF; local = cid; }
    else               { src = w2; dst = o2; Kt = DFF; Nt = D;   local = cid - E * 576; }
    const int chunksN = Nt / 128;             // W1: 24, W2: 6
    const int e  = local / 576;               // chunksK*chunksN == 576 for both
    const int c  = local % 576;
    const int ck = c / chunksN;               // k-chunk (32 rows)
    const int cn = c % chunksN;               // n-chunk (128 cols)

    const float* sbase = src + (size_t)e * Kt * Nt + (size_t)(ck * 32) * Nt + cn * 128;
    const int tid = threadIdx.x;

    // phase 1: 32x128 fp32 -> bf16 into LDS (coalesced float2 reads)
    for (int i = tid; i < (32 * 128) / 2; i += 256) {
        const int r  = i >> 6;                // 64 float2 per row
        const int cc = (i & 63) * 2;
        const float2 f = *(const float2*)(sbase + (size_t)r * Nt + cc);
        *(unsigned int*)(lt + r * 128 + cc) = pk2bf(f.x, f.y);
    }
    __syncthreads();

    // phase 2: each wave emits one 32x16 tile in fragment order (b128 stores)
    const int wave = tid >> 5, lane = tid & 31;
    const int kh = lane >> 4, nc = lane & 15;
    Frag f;
#pragma unroll
    for (int j = 0; j < 16; ++j)
        f.u[j] = lt[((j & 8) * 2 + kh * 8 + (j & 7)) * 128 + wave * 16 + nc];

    const size_t nt = (size_t)(cn * 8 + wave);    // global n-tile index
    unsigned short* obase = dst + (size_t)e * Kt * Nt
                          + (nt * (size_t)(Kt / 32) + ck) * 512 + lane * 16;
    *(uint4*)(obase)     = f.q[0];
    *(uint4*)(obase + 8) = f.q[1];
}

// ---------------------------------------------------------------------------
// K1: x_avg[b][d] = mean over T of x[b][t][d]   (mask is all-ones in reference)
// ---------------------------------------------------------------------------
__global__ void k_xavg(const float* __restrict__ x, float* __restrict__ xavg) {
    int idx = blockIdx.x * blockDim.x + threadIdx.x;   // < BZ*D
    int b = idx / D, d = idx - b * D;
    const float* p = x + b * (T * D) + d;
    float s = 0.f;
#pragma unroll
    for (int t = 0; t < T; ++t) s += p[t * D];
    xavg[idx] = s * (1.0f / (float)T);
}

// ---------------------------------------------------------------------------
// K2: gating: logits = x_avg @ Wg, softmax, top-2.
// ---------------------------------------------------------------------------
__global__ void k_gate(const float* __restrict__ xavg,
                       const float* __restrict__ wg,
                       float* __restrict__ scores_out,
                       float* __restrict__ route_out,
                       int* __restrict__ sel) {
    int b = threadIdx.x;                 // 0..BZ-1
    float l[E];
#pragma unroll
    for (int e = 0; e < E; ++e) l[e] = 0.f;
    const float* xa = xavg + b * D;
    for (int d = 0; d < D; ++d) {
        float v = xa[d];
        const float* w = wg + d * E;
#pragma unroll
        for (int e = 0; e < E; ++e) l[e] = fmaf(v, w[e], l[e]);
    }
    float m = l[0];
#pragma unroll
    for (int e = 1; e < E; ++e) m = fmaxf(m, l[e]);
    float s = 0.f;
#pragma unroll
    for (int e = 0; e < E; ++e) { l[e] = expf(l[e] - m); s += l[e]; }
    float inv = 1.0f / s;
#pragma unroll
    for (int e = 0; e < E; ++e) l[e] *= inv;

    // top-2, jax tie-break = lowest index first (strict >)
    int i1 = 0; float v1 = l[0];
#pragma unroll
    for (int e = 1; e < E; ++e) if (l[e] > v1) { v1 = l[e]; i1 = e; }
    int i2 = -1; float v2 = -1.0f;
#pragma unroll
    for (int e = 0; e < E; ++e) if (e != i1 && l[e] > v2) { v2 = l[e]; i2 = e; }

    scores_out[2 * b]     = v1;
    scores_out[2 * b + 1] = v2;
    route_out[2 * b]      = (float)i1;
    route_out[2 * b + 1]  = (float)i2;
    sel[2 * b]     = i1;
    sel[2 * b + 1] = i2;
}

// ---------------------------------------------------------------------------
// K3: fused expert FFN per beam-row. One block per row (256 blocks, 16 waves).
//   xb (32x768 bf16, 48 KB) and the full h (32x3072 bf16, 192 KB) live in LDS
//   (240 KB of the WGP's 320 KB). GEMM1 = x@W1 -> gelu -> h (LDS);
//   GEMM2 = h@W2 -> out. WMMA bf16 16x16x32.  B fragments are two
//   global_load_b128 from the pre-swizzled weights; A fragments are two
//   ds_load_b128 from LDS.  Each B fragment feeds 2 WMMAs (M=0 and M=16).
// ---------------------------------------------------------------------------
__global__ __launch_bounds__(512, 1)
void k_ffn(const float* __restrict__ x,
           const unsigned short* __restrict__ w1b,
           const unsigned short* __restrict__ w2b,
           const float* __restrict__ b1,
           const float* __restrict__ b2,
           const int* __restrict__ sel,
           float* __restrict__ out) {
    extern __shared__ char smem[];
    unsigned short* xb = (unsigned short*)smem;                 // T*D bf16
    unsigned short* hb = (unsigned short*)(smem + T * D * 2);   // T*DFF bf16

    const int row   = blockIdx.x;       // beam row 0..255
    const int batch = row >> 1;
    const int e     = sel[row];
    const int tid   = threadIdx.x;
    const int wave  = tid >> 5;
    const int lane  = tid & 31;
    const int mr    = lane & 15;        // row (A/C) within tile
    const int kh    = lane >> 4;        // lane-half selects K/M sub-range

    // Stage x row -> LDS bf16 (coalesced float4 loads, ds_store_b64)
    {
        const float4* xr = (const float4*)(x + (size_t)batch * (T * D));
        uint2* xo = (uint2*)xb;
        for (int i = tid; i < (T * D) / 4; i += 512) {
            float4 f = xr[i];
            uint2 o; o.x = pk2bf(f.x, f.y); o.y = pk2bf(f.z, f.w);
            xo[i] = o;
        }
    }
    __syncthreads();

    // ---- GEMM1 + bias + exact-form gelu -> hb ----
    const unsigned short* w1e = w1b + (size_t)e * D * DFF;
    const float* b1e = b1 + e * DFF;
    for (int nt = wave; nt < DFF / 16; nt += 16) {
        const int n0 = nt * 16;
        const unsigned short* btile = w1e + (size_t)nt * (D / 32) * 512 + lane * 16;
        v8f acc0 = {}; v8f acc1 = {};
#pragma unroll 2
        for (int kt = 0; kt < D / 32; ++kt) {
            const int k = kt * 32;
            Frag a0, a1, b;
            // B fragment: two coalesced b128 loads from swizzled weights
            const uint4* bp = (const uint4*)(btile + kt * 512);
            b.q[0] = bp[0];
            b.q[1] = bp[1];
            __builtin_prefetch(bp + 32, 0, 1);   // half a k-tile (512 B ahead)
            // A fragments (ISA 16-bit A 16x32 layout): two ds_load_b128 each
            a0.q[0] = *(const uint4*)(xb + mr * D + k + kh * 8);
            a0.q[1] = *(const uint4*)(xb + mr * D + k + 16 + kh * 8);
            a1.q[0] = *(const uint4*)(xb + (16 + mr) * D + k + kh * 8);
            a1.q[1] = *(const uint4*)(xb + (16 + mr) * D + k + 16 + kh * 8);
            acc0 = __builtin_amdgcn_wmma_f32_16x16x32_bf16(
                false, a0.v, false, b.v, (short)0, acc0, false, false);
            acc1 = __builtin_amdgcn_wmma_f32_16x16x32_bf16(
                false, a1.v, false, b.v, (short)0, acc1, false, false);
        }
        const float bias = b1e[n0 + mr];
#pragma unroll
        for (int i = 0; i < 8; ++i) {   // C layout: M = i + 8*kh, N = mr
            hb[(kh * 8 + i) * DFF + n0 + mr]      = f2bf(gelu_erf(acc0[i] + bias));
            hb[(16 + kh * 8 + i) * DFF + n0 + mr] = f2bf(gelu_erf(acc1[i] + bias));
        }
    }
    __syncthreads();

    // ---- GEMM2: out = h @ W2[e] + b2[e] ----
    const unsigned short* w2e = w2b + (size_t)e * DFF * D;
    const float* b2e = b2 + e * D;
    float* orow = out + (size_t)row * (T * D);
    for (int nt = wave; nt < D / 16; nt += 16) {
        const int n0 = nt * 16;
        const unsigned short* btile = w2e + (size_t)nt * (DFF / 32) * 512 + lane * 16;
        v8f acc0 = {}; v8f acc1 = {};
#pragma unroll 2
        for (int kt = 0; kt < DFF / 32; ++kt) {
            const int k = kt * 32;
            Frag a0, a1, b;
            const uint4* bp = (const uint4*)(btile + kt * 512);
            b.q[0] = bp[0];
            b.q[1] = bp[1];
            __builtin_prefetch(bp + 32, 0, 1);
            a0.q[0] = *(const uint4*)(hb + mr * DFF + k + kh * 8);
            a0.q[1] = *(const uint4*)(hb + mr * DFF + k + 16 + kh * 8);
            a1.q[0] = *(const uint4*)(hb + (16 + mr) * DFF + k + kh * 8);
            a1.q[1] = *(const uint4*)(hb + (16 + mr) * DFF + k + 16 + kh * 8);
            acc0 = __builtin_amdgcn_wmma_f32_16x16x32_bf16(
                false, a0.v, false, b.v, (short)0, acc0, false, false);
            acc1 = __builtin_amdgcn_wmma_f32_16x16x32_bf16(
                false, a1.v, false, b.v, (short)0, acc1, false, false);
        }
        const float bias = b2e[n0 + mr];
#pragma unroll
        for (int i = 0; i < 8; ++i) {
            orow[(kh * 8 + i) * D + n0 + mr]      = acc0[i] + bias;
            orow[(16 + kh * 8 + i) * D + n0 + mr] = acc1[i] + bias;
        }
    }
}

// ---------------------------------------------------------------------------
extern "C" void kernel_launch(void* const* d_in, const int* in_sizes, int n_in,
                              void* d_out, int out_size, void* d_ws, size_t ws_size,
                              hipStream_t stream) {
    (void)in_sizes; (void)n_in; (void)out_size; (void)ws_size;
    const float* x  = (const float*)d_in[0];
    // d_in[1] = attention_mask: reference overwrites it with ones -> unused
    const float* wg = (const float*)d_in[2];
    const float* w1 = (const float*)d_in[3];
    const float* b1 = (const float*)d_in[4];
    const float* w2 = (const float*)d_in[5];
    const float* b2 = (const float*)d_in[6];
    float* out = (float*)d_out;

    // workspace layout (needs ~76 MB)
    char* ws = (char*)d_ws;
    const size_t nw = (size_t)E * D * DFF;                  // 18874368 elems each
    unsigned short* w1bf = (unsigned short*)ws;             // nw bf16 (swizzled)
    unsigned short* w2bf = (unsigned short*)(ws + nw * 2);  // nw bf16 (swizzled)
    float* xavg = (float*)(ws + nw * 4);                    // BZ*D fp32
    int*   sel  = (int*)(ws + nw * 4 + (size_t)BZ * D * 4); // NROWS ints

    // output tail: out | beam_scores (256) | expert_route (256, as float)
    float* scores_out = out + (size_t)NROWS * T * D;
    float* route_out  = scores_out + NROWS;

    // K0: fp32 -> bf16 + B-fragment swizzle (2 * 8 * 576 chunks)
    k_swizzle<<<dim3(2 * E * 576), dim3(256), 0, stream>>>(w1, w2, w1bf, w2bf);

    // K1: mean over T  (BZ*D = 98304 threads -> 384 blocks)
    k_xavg<<<dim3((BZ * D) / 256), dim3(256), 0, stream>>>(x, xavg);

    // K2: gating + top-2 + tail outputs
    k_gate<<<dim3(1), dim3(BZ), 0, stream>>>(xavg, wg, scores_out, route_out, sel);

    // K3: fused expert FFN (240 KB dynamic LDS of the WGP's 320 KB)
    const int lds_bytes = (T * D + T * DFF) * 2;  // 245760
    hipFuncSetAttribute(reinterpret_cast<const void*>(k_ffn),
                        hipFuncAttributeMaxDynamicSharedMemorySize, lds_bytes);
    k_ffn<<<dim3(NROWS), dim3(512), lds_bytes, stream>>>(
        x, w1bf, w2bf, b1, b2, sel, out);
}
